// adaptiveFusion_23441931501709
// MI455X (gfx1250) — matrix operand order repaired and checked by reading, not compiled, not run on hardware
//
#include <hip/hip_runtime.h>

// MI455X / gfx1250, wave32. bf16 WMMA 16x16x32, f32 accumulation.
// One workgroup (8 waves) per (b,n) attention head; whole block fused in 288KB LDS.
// Weight GEMMs: wave w owns output columns [32w,32w+32); the 16 weight B-fragments
// (128 VGPRs) are loaded from L2 ONCE per GEMM and reused across all 6 row tiles.
// All helpers are force-inlined + template-specialized so LDS pointers stay in
// address space 3 (no generic-pointer null checks) and no dead pointer params live.

typedef __attribute__((ext_vector_type(16))) __bf16 v16bf;
typedef __attribute__((ext_vector_type(8)))  float  v8f;

#define TT 96
#define FF 256
#define NN 400
#define BB 8

static __device__ __forceinline__ unsigned short f2bf(float f) {
  unsigned int u = __float_as_uint(f);
  u += 0x7FFFu + ((u >> 16) & 1u);     // round-to-nearest-even
  return (unsigned short)(u >> 16);
}

union FragU { v16bf v; uint4 q[2]; unsigned int u[8]; };

// A fragment (16x32 bf16). elem(m,k) = base[(m0+m)*ld + k0+k]
// Per lane: regs 0..7 = K[16h .. 16h+15]
// -> 32B contiguous -> exactly 2x b128 loads.
static __device__ __forceinline__ v16bf frag_a(const unsigned short* base, int ld,
                                               int m0, int k0, int lane) {
  FragU f;
  const int m = m0 + (lane & 15), h = lane >> 4;
  const unsigned short* p = base + m * ld + k0 + 16 * h;
  f.q[0] = *(const uint4*)(p);
  f.q[1] = *(const uint4*)(p + 8);
  return f.v;
}

// B fragment (32x16 bf16), elem(k,n) = base[(n0+n)*ld + k0+k]   (B = Mem^T, Mem [N][K])
// Per lane: regs r hold K = k0 + 16h + 2r (+1) -> 32B contiguous -> 2x b128 loads.
static __device__ __forceinline__ v16bf frag_b_nk(const unsigned short* base, int ld,
                                                  int n0, int k0, int lane) {
  FragU f;
  const int n = n0 + (lane & 15), h = lane >> 4;
  const unsigned short* p = base + n * ld + k0 + 16 * h;
  f.q[0] = *(const uint4*)(p);
  f.q[1] = *(const uint4*)(p + 8);
  return f.v;
}

#define WMMA_BF16(a, b, c) \
  __builtin_amdgcn_wmma_f32_16x16x32_bf16(false, (a), false, (b), (short)0, (c), false, false)

// C[96,256] = act(A[96,256] @ W[256,256]^T + bias). W row-major [N][K] bf16 in L2.
// Wave w owns columns [32w, 32w+32); weights resident in 128 VGPRs across row tiles.
// F32OUT: write f32 to outp (stride FF). Else bf16 (optionally transposed, stride TT).
template <bool RELU, bool TRANS, bool F32OUT>
static __device__ __forceinline__ void gemm_w(const unsigned short* __restrict__ A,
                                              const unsigned short* __restrict__ W,
                                              const float* __restrict__ bias,
                                              void* __restrict__ outp,
                                              int wave, int lane) {
  const int n = lane & 15, h = lane >> 4;
  const int ni = wave * 32;
  v16bf wf0[8], wf1[8];
#pragma unroll
  for (int kk = 0; kk < 8; ++kk) {
    wf0[kk] = frag_b_nk(W, FF, ni,      kk * 32, lane);
    wf1[kk] = frag_b_nk(W, FF, ni + 16, kk * 32, lane);
  }
  const float b0i = bias[ni + n];
  const float b1i = bias[ni + 16 + n];
  for (int mt = 0; mt < 6; ++mt) {       // rolled: bounds I-cache, C regs recycled
    const int mi = mt * 16;
    v8f c0, c1;
#pragma unroll
    for (int r = 0; r < 8; ++r) { c0[r] = b0i; c1[r] = b1i; }
#pragma unroll
    for (int kk = 0; kk < 8; ++kk) {
      v16bf a = frag_a(A, FF, mi, kk * 32, lane);
      c0 = WMMA_BF16(a, wf0[kk], c0);
      c1 = WMMA_BF16(a, wf1[kk], c1);
    }
#pragma unroll
    for (int r = 0; r < 8; ++r) {        // D: row = r + 8*(lane>>4), col = lane&15
      const int row = mi + r + 8 * h;
      float v0 = c0[r], v1 = c1[r];
      if (RELU) { v0 = fmaxf(v0, 0.f); v1 = fmaxf(v1, 0.f); }
      if constexpr (F32OUT) {
        float* o = (float*)outp;
        o[row * FF + ni + n]      = v0;
        o[row * FF + ni + 16 + n] = v1;
      } else if constexpr (TRANS) {      // store V as VT[256][96]
        unsigned short* o = (unsigned short*)outp;
        o[(ni + n) * TT + row]      = f2bf(v0);
        o[(ni + 16 + n) * TT + row] = f2bf(v1);
      } else {
        unsigned short* o = (unsigned short*)outp;
        o[row * FF + ni + n]      = f2bf(v0);
        o[row * FF + ni + 16 + n] = f2bf(v1);
      }
    }
  }
}

// scores[96,96] = Q[96,256] @ K[96,256]^T  (both LDS), f32 out.
static __device__ __forceinline__ void gemm_scores(const unsigned short* __restrict__ Q,
                                                   const unsigned short* __restrict__ K,
                                                   float* __restrict__ S, int wave, int lane) {
  const int n = lane & 15, h = lane >> 4;
  for (int t = wave; t < 18; t += 8) {   // 6 row tiles x 3 col pairs
    const int mi = (t / 3) * 16, ni = (t % 3) * 32;
    v8f c0, c1;
#pragma unroll
    for (int r = 0; r < 8; ++r) { c0[r] = 0.f; c1[r] = 0.f; }
#pragma unroll
    for (int kk = 0; kk < 8; ++kk) {
      v16bf a  = frag_a(Q, FF, mi, kk * 32, lane);
      v16bf b0 = frag_b_nk(K, FF, ni,      kk * 32, lane);
      v16bf b1 = frag_b_nk(K, FF, ni + 16, kk * 32, lane);
      c0 = WMMA_BF16(a, b0, c0);
      c1 = WMMA_BF16(a, b1, c1);
    }
#pragma unroll
    for (int r = 0; r < 8; ++r) {
      const int row = mi + r + 8 * h;
      S[row * TT + ni + n]      = c0[r];
      S[row * TT + ni + 16 + n] = c1[r];
    }
  }
}

// out[96,256] bf16 = attn[96,96]bf16 @ V; V held transposed VT[256][96].
// Wave w owns feature columns [32w, 32w+32); V fragments resident in 48 VGPRs.
static __device__ __forceinline__ void gemm_av(const unsigned short* __restrict__ attn,
                                               const unsigned short* __restrict__ VT,
                                               unsigned short* __restrict__ outb,
                                               int wave, int lane) {
  const int n = lane & 15, h = lane >> 4;
  const int ni = wave * 32;
  v16bf v0[3], v1[3];
#pragma unroll
  for (int kk = 0; kk < 3; ++kk) {
    v0[kk] = frag_b_nk(VT, TT, ni,      kk * 32, lane);
    v1[kk] = frag_b_nk(VT, TT, ni + 16, kk * 32, lane);
  }
  for (int mt = 0; mt < 6; ++mt) {
    const int mi = mt * 16;
    v8f c0, c1;
#pragma unroll
    for (int r = 0; r < 8; ++r) { c0[r] = 0.f; c1[r] = 0.f; }
#pragma unroll
    for (int kk = 0; kk < 3; ++kk) {
      v16bf a = frag_a(attn, 2 * TT, mi, kk * 32, lane);   // attn stride 192 ushorts
      c0 = WMMA_BF16(a, v0[kk], c0);
      c1 = WMMA_BF16(a, v1[kk], c1);
    }
#pragma unroll
    for (int r = 0; r < 8; ++r) {
      const int row = mi + r + 8 * h;
      outb[row * FF + ni + n]      = f2bf(c0[r]);
      outb[row * FF + ni + 16 + n] = f2bf(c1[r]);
    }
  }
}

// y = LayerNorm(x1 + x2), rows of 256. Wave-per-row, float4 I/O, shuffle reduce.
// GLOBAL: write f32 to gout (coalesced b128). Else write f32 LDS + packed bf16 LDS.
template <bool GLOBAL>
static __device__ __forceinline__ void add_ln(const float* __restrict__ x1,
                                              const float* __restrict__ x2,
                                              float* __restrict__ outf,
                                              unsigned short* __restrict__ outb,
                                              float* __restrict__ gout, long gbase,
                                              int wave, int lane) {
  for (int t = wave; t < TT; t += 8) {
    const float4* p1 = (const float4*)(x1 + t * FF);
    const float4* p2 = (const float4*)(x2 + t * FF);
    float4 u[2];
    float s = 0.f;
#pragma unroll
    for (int j = 0; j < 2; ++j) {
      const float4 a = p1[lane + 32 * j], b = p2[lane + 32 * j];
      float4 v; v.x = a.x + b.x; v.y = a.y + b.y; v.z = a.z + b.z; v.w = a.w + b.w;
      u[j] = v;
      s += (v.x + v.y) + (v.z + v.w);
    }
#pragma unroll
    for (int off = 16; off >= 1; off >>= 1) s += __shfl_xor(s, off, 32);
    const float mean = s * (1.f / 256.f);
    float var = 0.f;
#pragma unroll
    for (int j = 0; j < 2; ++j) {
      const float4 v = u[j];
      var += (v.x - mean) * (v.x - mean) + (v.y - mean) * (v.y - mean)
           + (v.z - mean) * (v.z - mean) + (v.w - mean) * (v.w - mean);
    }
#pragma unroll
    for (int off = 16; off >= 1; off >>= 1) var += __shfl_xor(var, off, 32);
    const float rs = rsqrtf(var * (1.f / 256.f) + 1e-5f);
#pragma unroll
    for (int j = 0; j < 2; ++j) {
      const float4 v = u[j];
      float4 y; y.x = (v.x - mean) * rs; y.y = (v.y - mean) * rs;
      y.z = (v.z - mean) * rs; y.w = (v.w - mean) * rs;
      if constexpr (GLOBAL) {
        ((float4*)(gout + gbase + (long)t * (NN * FF)))[lane + 32 * j] = y;
      } else {
        ((float4*)(outf + t * FF))[lane + 32 * j] = y;
        uint2 p;
        p.x = (unsigned)f2bf(y.x) | ((unsigned)f2bf(y.y) << 16);
        p.y = (unsigned)f2bf(y.z) | ((unsigned)f2bf(y.w) << 16);
        ((uint2*)(outb + t * FF))[lane + 32 * j] = p;
      }
    }
  }
}

__global__ void convert_w_kernel(const float* __restrict__ src,
                                 unsigned short* __restrict__ dst, int n) {
  const int i = blockIdx.x * blockDim.x + threadIdx.x;
  if (i < n) dst[i] = f2bf(src[i]);
}

__global__ __launch_bounds__(256)
void fused_attn_block(const float* __restrict__ xl, const float* __restrict__ xh,
                      const float* __restrict__ te,
                      const unsigned short* __restrict__ Wq, const unsigned short* __restrict__ Wk,
                      const unsigned short* __restrict__ Wv, const unsigned short* __restrict__ Wo,
                      const unsigned short* __restrict__ Wf1, const unsigned short* __restrict__ Wf2,
                      const float* __restrict__ bq, const float* __restrict__ bk,
                      const float* __restrict__ bv, const float* __restrict__ bo,
                      const float* __restrict__ bf1, const float* __restrict__ bf2,
                      float* __restrict__ out) {
  // LDS plan (288 KB): sF f32[96*256] | T0 bf16 | T2 bf16 | T1 bf16 | T3 bf16 (48KB each)
  // scores f32[96*96] overlays T1; 96KB f32 GEMM temp overlays T1+T3 (adjacent).
  __shared__ __align__(16) unsigned char smem[294912];
  float*          sF = (float*)smem;
  unsigned short* T0 = (unsigned short*)(smem + 98304);
  unsigned short* T2 = (unsigned short*)(smem + 98304 + 49152);
  unsigned short* T1 = (unsigned short*)(smem + 98304 + 2 * 49152);
  unsigned short* T3 = (unsigned short*)(smem + 98304 + 3 * 49152);
  float*          S  = (float*)(smem + 98304 + 2 * 49152);   // 96x96 scores
  float*         TMP = S;                                     // 96x256 f32 temp

  const int tid = threadIdx.x, wave = tid >> 5, lane = tid & 31;
  const int b = blockIdx.x / NN, n = blockIdx.x % NN;
  const long bt0 = (long)b * TT;

  // Phase 0: stream rows in as float4, add te, keep f32 residual + packed bf16.
  {
    const float4* xl4 = (const float4*)xl;
    const float4* xh4 = (const float4*)xh;
    const float4* te4 = (const float4*)te;
    float4* sF4 = (float4*)sF;
    uint2* T0p = (uint2*)T0;
    uint2* T1p = (uint2*)T1;
    for (int i = tid; i < TT * 64; i += 256) {
      const int t = i >> 6, f4 = i & 63;
      const long g = ((bt0 + t) * NN + n) * 64 + f4;
      const float4 tv = te4[(bt0 + t) * 64 + f4];
      float4 a = xl4[g], hh = xh4[g];
      a.x += tv.x; a.y += tv.y; a.z += tv.z; a.w += tv.w;
      hh.x += tv.x; hh.y += tv.y; hh.z += tv.z; hh.w += tv.w;
      sF4[i] = a;
      uint2 pa, ph;
      pa.x = (unsigned)f2bf(a.x)  | ((unsigned)f2bf(a.y)  << 16);
      pa.y = (unsigned)f2bf(a.z)  | ((unsigned)f2bf(a.w)  << 16);
      ph.x = (unsigned)f2bf(hh.x) | ((unsigned)f2bf(hh.y) << 16);
      ph.y = (unsigned)f2bf(hh.z) | ((unsigned)f2bf(hh.w) << 16);
      T0p[i] = pa; T1p[i] = ph;
    }
  }
  __syncthreads();

  gemm_w<false, false, false>(T0, Wq, bq, T2, wave, lane);     // Q  -> T2
  __syncthreads();
  gemm_w<true,  false, false>(T1, Wk, bk, T0, wave, lane);     // K  -> T0 (relu)
  __syncthreads();
  gemm_w<true,  true,  false>(T1, Wv, bv, T3, wave, lane);     // VT -> T3 (relu, transposed)
  __syncthreads();

  gemm_scores(T2, T0, S, wave, lane);                          // scores -> S (over dead xh')
  __syncthreads();

  // Causal softmax, wave-per-row; masked entries underflow to exactly 0
  // (reference NEG/16 ~ -2048 does the same in fp32). attn written bf16 in place.
  for (int t = wave; t < TT; t += 8) {
    float* row = S + t * TT;
    const float sc = 1.f / 16.f;                               // 1/sqrt(256)
    float x0 = (lane      <= t) ? row[lane]      * sc : -1e30f;
    float x1 = (lane + 32 <= t) ? row[lane + 32] * sc : -1e30f;
    float x2 = (lane + 64 <= t) ? row[lane + 64] * sc : -1e30f;
    float mx = fmaxf(x0, fmaxf(x1, x2));
#pragma unroll
    for (int off = 16; off >= 1; off >>= 1) mx = fmaxf(mx, __shfl_xor(mx, off, 32));
    const float e0 = __expf(x0 - mx), e1 = __expf(x1 - mx), e2 = __expf(x2 - mx);
    float s = e0 + e1 + e2;
#pragma unroll
    for (int off = 16; off >= 1; off >>= 1) s += __shfl_xor(s, off, 32);
    const float inv = 1.f / s;
    unsigned short* arow = (unsigned short*)row;               // bf16, row stride 192
    arow[lane]      = f2bf(e0 * inv);
    arow[lane + 32] = f2bf(e1 * inv);
    arow[lane + 64] = f2bf(e2 * inv);
  }
  __syncthreads();

  gemm_av((const unsigned short*)S, T3, T2, wave, lane);       // attn @ V -> T2
  __syncthreads();

  gemm_w<false, false, true>(T2, Wo, bo, TMP, wave, lane);     // Wo -> f32 TMP
  __syncthreads();

  add_ln<false>(TMP, sF, sF, T0, nullptr, 0, wave, lane);      // out1 -> sF (f32) + T0 (bf16)
  __syncthreads();

  gemm_w<true,  false, false>(T0, Wf1, bf1, T2, wave, lane);   // FFN1 -> T2 (relu)
  __syncthreads();
  gemm_w<false, false, true>(T2, Wf2, bf2, TMP, wave, lane);   // FFN2 -> f32 TMP
  __syncthreads();

  add_ln<true>(TMP, sF, nullptr, nullptr, out,
               (long)b * TT * NN * FF + (long)n * FF, wave, lane);  // final LN -> global
}

extern "C" void kernel_launch(void* const* d_in, const int* in_sizes, int n_in,
                              void* d_out, int out_size, void* d_ws, size_t ws_size,
                              hipStream_t stream) {
  (void)in_sizes; (void)n_in; (void)out_size; (void)ws_size;   // needs 768 KB ws
  const float* xl  = (const float*)d_in[0];
  const float* xh  = (const float*)d_in[1];
  const float* te  = (const float*)d_in[2];
  const float* Wq  = (const float*)d_in[3];
  const float* bq  = (const float*)d_in[4];
  const float* Wk  = (const float*)d_in[5];
  const float* bk  = (const float*)d_in[6];
  const float* Wv  = (const float*)d_in[7];
  const float* bv  = (const float*)d_in[8];
  const float* Wo  = (const float*)d_in[9];
  const float* bo  = (const float*)d_in[10];
  const float* Wf1 = (const float*)d_in[11];
  const float* bf1 = (const float*)d_in[12];
  const float* Wf2 = (const float*)d_in[13];
  const float* bf2 = (const float*)d_in[14];

  // Pre-convert the six 256x256 weight matrices to bf16 in d_ws (L2-resident).
  const int WSZ = FF * FF;
  unsigned short* wsb = (unsigned short*)d_ws;
  const float* Ws[6] = {Wq, Wk, Wv, Wo, Wf1, Wf2};
  for (int i = 0; i < 6; ++i)
    convert_w_kernel<<<(WSZ + 255) / 256, 256, 0, stream>>>(Ws[i], wsb + (size_t)i * WSZ, WSZ);

  fused_attn_block<<<BB * NN, 256, 0, stream>>>(
      xl, xh, te,
      wsb, wsb + WSZ, wsb + 2 * WSZ, wsb + 3 * WSZ, wsb + 4 * WSZ, wsb + 5 * WSZ,
      bq, bk, bv, bo, bf1, bf2, (float*)d_out);
}
